// FineGrainedOpLSTMCell_v1_47399259079383
// MI455X (gfx1250) — matrix operand order repaired and checked by reading, not compiled, not run on hardware
//
#include <hip/hip_runtime.h>

typedef __bf16 bf16_t;
typedef __attribute__((ext_vector_type(16))) __bf16 v16bf;
typedef __attribute__((ext_vector_type(8)))  float  v8f;
typedef __attribute__((ext_vector_type(4)))  float  f32x4;

#define BATCH   131072
#define HDIM    128
#define KDIM    256          // concat [x | h_prev]
#define MT      32           // batch rows per block
#define A_STRIDE 264         // 256 + 8 pad bf16 (528B rows, 16B aligned, bank-conflict-free)
#define G_STRIDE 132         // 128 + 4 pad f32

// ---------------------------------------------------------------------------
// Prep: pack [W;U] (K=256 x N=128, fp32) per gate into WMMA-B fragment order,
// bf16. Fragment tile = 32(K) x 16(N). Per-lane 16 contiguous bf16 (32B):
// lane l -> column n = l%16; lanes 0-15 hold K=kt*32+0..15, lanes 16-31 hold
// K=kt*32+16..31 (element j of v16bf <-> K offset j within the half).
// ---------------------------------------------------------------------------
__global__ __launch_bounds__(256) void lstm_pack_weights(
    const float* __restrict__ Wi, const float* __restrict__ Ui,
    const float* __restrict__ Wf, const float* __restrict__ Uf,
    const float* __restrict__ Wg, const float* __restrict__ Ug,
    const float* __restrict__ Wo, const float* __restrict__ Uo,
    bf16_t* __restrict__ packed)
{
    const float* Wt[4] = {Wi, Wf, Wg, Wo};
    const float* Ut[4] = {Ui, Uf, Ug, Uo};
    int tid = blockIdx.x * 256 + threadIdx.x;
    if (tid >= 4 * 8 * 8 * 32 * 16) return;
    int e    =  tid        & 15;   // element within lane chunk
    int lane = (tid >> 4)  & 31;
    int nt   = (tid >> 9)  & 7;    // N tile
    int kt   = (tid >> 12) & 7;    // K tile
    int g    = (tid >> 15) & 3;    // gate: i,f,g,o
    int klocal = ((lane >> 4) << 4) + e;          // 0..31
    int k = kt * 32 + klocal;
    int n = nt * 16 + (lane & 15);
    float v = (k < HDIM) ? Wt[g][k * HDIM + n] : Ut[g][(k - HDIM) * HDIM + n];
    packed[tid] = (bf16_t)v;
}

__device__ __forceinline__ float fast_sigmoid(float x) {
    return 1.0f / (1.0f + exp2f(-1.442695040888963f * x));
}
__device__ __forceinline__ float fast_tanh(float x) {
    return 2.0f / (1.0f + exp2f(-2.885390081777927f * x)) - 1.0f;
}

// ---------------------------------------------------------------------------
// Main fused LSTM cell: 8 waves per block, wave = (row-strip s in {0,1},
// gate g in {i,f,g,o}), each wave does 64x v_wmma_f32_16x16x32_bf16.
// ---------------------------------------------------------------------------
__global__ __launch_bounds__(256) void lstm_cell_wmma(
    const float* __restrict__ x,     const float* __restrict__ hprev,
    const float* __restrict__ cprev,
    const float* __restrict__ bi,    const float* __restrict__ bfv,
    const float* __restrict__ bg,    const float* __restrict__ bo,
    const bf16_t* __restrict__ packed,
    float* __restrict__ hout,        float* __restrict__ cout)
{
    __shared__ bf16_t Atile[MT][A_STRIDE];       // ~16.9 KB
    __shared__ float  Gates[4][MT][G_STRIDE];    // ~67.6 KB
    __shared__ float  Bias[4][HDIM];             //   2.0 KB

    const int tid  = threadIdx.x;
    const int row0 = blockIdx.x * MT;

    // ---- stage A = [x | h_prev] tile into LDS as bf16 ----
    {
        int r  = tid >> 3;            // 0..31
        int cb = (tid & 7) * 16;      // 0..112
        const f32x4* xr = (const f32x4*)(x     + (size_t)(row0 + r) * HDIM + cb);
        const f32x4* hr = (const f32x4*)(hprev + (size_t)(row0 + r) * HDIM + cb);
#pragma unroll
        for (int j = 0; j < 4; ++j) {
            f32x4 vx = xr[j];
            f32x4 vh = hr[j];
            int c = cb + j * 4;
            Atile[r][c + 0] = (bf16_t)vx.x;  Atile[r][c + 1] = (bf16_t)vx.y;
            Atile[r][c + 2] = (bf16_t)vx.z;  Atile[r][c + 3] = (bf16_t)vx.w;
            Atile[r][128 + c + 0] = (bf16_t)vh.x;  Atile[r][128 + c + 1] = (bf16_t)vh.y;
            Atile[r][128 + c + 2] = (bf16_t)vh.z;  Atile[r][128 + c + 3] = (bf16_t)vh.w;
        }
        const float* bp[4] = {bi, bfv, bg, bo};
#pragma unroll
        for (int i = tid; i < 4 * HDIM; i += 256)
            Bias[i >> 7][i & 127] = bp[i >> 7][i & 127];
    }
    // warm the L2 path for the packed weights (global_prefetch_b8)
    __builtin_prefetch(packed + (size_t)(tid & 3) * 64 * 512, 0, 1);
    __syncthreads();

    // ---- WMMA main loop ----
    const int lane = tid & 31;
    const int wave = tid >> 5;         // 0..7
    const int g    = wave & 3;         // gate
    const int s    = wave >> 2;        // row strip 0..1
    const int half = lane >> 4;
    const int mrow = s * 16 + (lane & 15);

    v8f acc[8] = {};

#pragma unroll
    for (int kt = 0; kt < 8; ++kt) {
        // A fragment (16x32 bf16): lane holds row mrow, K
        // {kt*32 + half*16 + 0..15}
        union { f32x4 q[2]; v16bf v; } afrag;
        const bf16_t* ap = &Atile[mrow][kt * 32 + half * 16];
        afrag.q[0] = *(const f32x4*)(ap);
        afrag.q[1] = *(const f32x4*)(ap + 8);

        const bf16_t* bptr = packed + (size_t)((g * 8 + kt) * 8) * 512 + lane * 16;
#pragma unroll
        for (int nt = 0; nt < 8; ++nt) {
            union { f32x4 q[2]; v16bf v; } bfrag;
            bfrag.q[0] = *(const f32x4*)(bptr);
            bfrag.q[1] = *(const f32x4*)(bptr + 8);
            bptr += 512;
            acc[nt] = __builtin_amdgcn_wmma_f32_16x16x32_bf16(
                false, afrag.v, false, bfrag.v, (short)0, acc[nt], false, false);
        }
    }

    // ---- spill gate pre-activations to LDS for cross-wave combine ----
    // C layout: VGPR r, lanes 0-15 -> M=r, lanes 16-31 -> M=r+8; N = lane%16
    {
        const int ccol  = lane & 15;
        const int rbase = half * 8;
#pragma unroll
        for (int nt = 0; nt < 8; ++nt)
#pragma unroll
            for (int r = 0; r < 8; ++r)
                Gates[g][s * 16 + rbase + r][nt * 16 + ccol] = acc[nt][r];
    }
    __syncthreads();

    // ---- fused elementwise epilogue ----
    {
        int r  = tid >> 3;            // 0..31
        int cb = (tid & 7) * 16;
        size_t gidx = (size_t)(row0 + r) * HDIM + cb;
#pragma unroll
        for (int j = 0; j < 4; ++j) {
            int c = cb + j * 4;
            f32x4 pi = *(const f32x4*)&Gates[0][r][c];
            f32x4 pf = *(const f32x4*)&Gates[1][r][c];
            f32x4 pg = *(const f32x4*)&Gates[2][r][c];
            f32x4 po = *(const f32x4*)&Gates[3][r][c];
            f32x4 bi4 = *(const f32x4*)&Bias[0][c];
            f32x4 bf4 = *(const f32x4*)&Bias[1][c];
            f32x4 bg4 = *(const f32x4*)&Bias[2][c];
            f32x4 bo4 = *(const f32x4*)&Bias[3][c];
            f32x4 cp  = *(const f32x4*)(cprev + gidx + j * 4);
            f32x4 c4, h4;
#define LSTM_ELEM(K) { \
            float ig = fast_sigmoid(pi.K + bi4.K); \
            float fg = fast_sigmoid(pf.K + bf4.K); \
            float gg = fast_tanh   (pg.K + bg4.K); \
            float og = fast_sigmoid(po.K + bo4.K); \
            float cn = fg * cp.K + ig * gg; \
            c4.K = cn; h4.K = og * fast_tanh(cn); }
            LSTM_ELEM(x) LSTM_ELEM(y) LSTM_ELEM(z) LSTM_ELEM(w)
#undef LSTM_ELEM
            *(f32x4*)(cout + gidx + j * 4) = c4;
            *(f32x4*)(hout + gidx + j * 4) = h4;
        }
    }
}

extern "C" void kernel_launch(void* const* d_in, const int* in_sizes, int n_in,
                              void* d_out, int out_size, void* d_ws, size_t ws_size,
                              hipStream_t stream) {
    const float* x  = (const float*)d_in[0];
    const float* hp = (const float*)d_in[1];
    const float* cp = (const float*)d_in[2];
    const float* Wi = (const float*)d_in[3];
    const float* Ui = (const float*)d_in[4];
    const float* bi = (const float*)d_in[5];
    const float* Wf = (const float*)d_in[6];
    const float* Uf = (const float*)d_in[7];
    const float* bf = (const float*)d_in[8];
    const float* Wg = (const float*)d_in[9];
    const float* Ug = (const float*)d_in[10];
    const float* bg = (const float*)d_in[11];
    const float* Wo = (const float*)d_in[12];
    const float* Uo = (const float*)d_in[13];
    const float* bo = (const float*)d_in[14];
    float* out = (float*)d_out;
    bf16_t* packed = (bf16_t*)d_ws;   // 128 KB fragment-packed bf16 weights

    lstm_pack_weights<<<512, 256, 0, stream>>>(Wi, Ui, Wf, Uf, Wg, Ug, Wo, Uo, packed);
    lstm_cell_wmma<<<BATCH / MT, 256, 0, stream>>>(
        x, hp, cp, bi, bf, bg, bo, packed,
        out, out + (size_t)BATCH * HDIM);
}
